// Repeat_Explore_Mechanism_28716151341088
// MI455X (gfx1250) — matrix operand order repaired and checked by reading, not compiled, not run on hardware
//
#include <hip/hip_runtime.h>
#include <stdint.h>

#define Bsz  2048
#define Ssz  200
#define Hsz  512
#define AROWS 208
#define ASTR  520                      // halves per LDS row (pad -> conflict-free frag loads)
#define A_BYTES (AROWS * ASTR * 2)     // 216320
#define SMEM_BYTES (A_BYTES + 256*4 + 32*4)

typedef __attribute__((ext_vector_type(16))) __bf16 v16bf;
typedef __attribute__((ext_vector_type(8)))  float  v8f;

union Frag { v16bf v; uint4 q[2]; };

__device__ __forceinline__ uint16_t f2bf(float f) {
    union { float f; uint32_t u; } a; a.f = f;
    uint32_t u = a.u + 0x7FFFu + ((a.u >> 16) & 1u);   // round-to-nearest-even
    return (uint16_t)(u >> 16);
}
__device__ __forceinline__ float bf2f(uint32_t h) {
    union { uint32_t u; float f; } a; a.u = h << 16; return a.f;
}
__device__ __forceinline__ uint32_t pack2(float a, float b) {
    return (uint32_t)f2bf(a) | ((uint32_t)f2bf(b) << 16);
}

// ---- fragment loaders --------------------------------------------------
// A 16x32 bf16: lanes 0-15 row=lane, K 0..7 & 16..23 ; lanes 16-31 row=lane-16, K 8..15 & 24..31
__device__ __forceinline__ v16bf load_a_lds(const uint16_t* A, int rowbase, int kbase, int lane) {
    const uint16_t* p = A + (rowbase + (lane & 15)) * ASTR + kbase + ((lane >> 4) << 3);
    Frag f;
    f.q[0] = *(const uint4*)p;          // 16B: first K chunk
    f.q[1] = *(const uint4*)(p + 16);   // 16B: +16 K chunk
    return f.v;
}
__device__ __forceinline__ v16bf load_a_glb(const uint16_t* A, int rowbase, int kbase, int lane) {
    const uint16_t* p = A + (size_t)(rowbase + (lane & 15)) * Hsz + kbase + ((lane >> 4) << 3);
    Frag f;
    f.q[0] = *(const uint4*)p;
    f.q[1] = *(const uint4*)(p + 16);
    return f.v;
}
// B 32x16 bf16: lane = output column n = weight row; lanes 0-15 K 0..7 & 16..23, lanes 16-31 K 8..15 & 24..31
// (K contiguous along the row-major weight row W[n][k])
__device__ __forceinline__ v16bf load_b_glb(const uint16_t* W, int nbase, int kbase, int lane) {
    const uint16_t* p = W + (size_t)(nbase + (lane & 15)) * Hsz + kbase + ((lane >> 4) << 3);
    Frag f;
    f.q[0] = *(const uint4*)p;
    f.q[1] = *(const uint4*)(p + 16);
    return f.v;
}

// ---- prep kernels ------------------------------------------------------
__global__ void convert_bf16(const float* __restrict__ src, uint16_t* __restrict__ dst, int n) {
    int i = blockIdx.x * blockDim.x + threadIdx.x;
    int stride = gridDim.x * blockDim.x;
    for (; i < n; i += stride) dst[i] = f2bf(src[i]);
}

// l = last_memory @ Wr^T  : M=2048, N=512, K=512 (bf16 WMMA, f32 acc)
__global__ void lgemm_kernel(const uint16_t* __restrict__ LM, const uint16_t* __restrict__ Wr,
                             float* __restrict__ Lout) {
    int lane  = threadIdx.x & 31;
    int mt    = blockIdx.x >> 3;   // 0..127 (batch tiles)
    int ngrp  = blockIdx.x & 7;    // 0..7   (64 output cols each)
    int mbase = mt * 16;

    v8f zero = {};
    v8f acc[4] = {zero, zero, zero, zero};
    for (int k = 0; k < Hsz; k += 32) {
        v16bf a = load_a_glb(LM, mbase, k, lane);
#pragma unroll
        for (int nt = 0; nt < 4; ++nt) {
            v16bf bf = load_b_glb(Wr, ngrp * 64 + nt * 16, k, lane);
            acc[nt] = __builtin_amdgcn_wmma_f32_16x16x32_bf16(
                false, a, false, bf, (short)0, acc[nt], false, false);
        }
    }
    int rowadd = (lane >> 4) << 3;
#pragma unroll
    for (int nt = 0; nt < 4; ++nt) {
        int col = ngrp * 64 + nt * 16 + (lane & 15);
#pragma unroll
        for (int r = 0; r < 8; ++r)
            Lout[(size_t)(mbase + r + rowadd) * Hsz + col] = acc[nt][r];
    }
}

// ---- one fully-unrolled GEMM job: MCOUNT m-tiles x 2 n-tiles ----------
template <int MCOUNT>
__device__ __forceinline__ void gemm_job(
    const uint16_t* __restrict__ Alds, const uint16_t* __restrict__ Ur,
    const float* __restrict__ lrow, const float* __restrict__ Vre,
    float* __restrict__ scores, int mg, int ng, int lane)
{
    v8f zero = {};
    v8f acc[MCOUNT][2];
#pragma unroll
    for (int mt = 0; mt < MCOUNT; ++mt) { acc[mt][0] = zero; acc[mt][1] = zero; }

    for (int k = 0; k < Hsz; k += 32) {
        v16bf afr[MCOUNT];
#pragma unroll
        for (int mt = 0; mt < MCOUNT; ++mt)
            afr[mt] = load_a_lds(Alds, (mg * 4 + mt) * 16, k, lane);
#pragma unroll
        for (int nt = 0; nt < 2; ++nt) {
            v16bf bfr = load_b_glb(Ur, ng * 32 + nt * 16, k, lane);
#pragma unroll
            for (int mt = 0; mt < MCOUNT; ++mt)
                acc[mt][nt] = __builtin_amdgcn_wmma_f32_16x16x32_bf16(
                    false, afr[mt], false, bfr, (short)0, acc[mt][nt], false, false);
        }
    }
    // epilogue: h = tanh(a + l), score += h * Vre  (h never stored)
    int n0 = ng * 32 + (lane & 15);
    int n1 = n0 + 16;
    float lv0 = lrow[n0], lv1 = lrow[n1];
    float vr0 = Vre[n0],  vr1 = Vre[n1];
    int rowadd = (lane >> 4) << 3;
#pragma unroll
    for (int mt = 0; mt < MCOUNT; ++mt) {
        int rowbase = (mg * 4 + mt) * 16 + rowadd;
#pragma unroll
        for (int r = 0; r < 8; ++r) {
            float v = tanhf(acc[mt][0][r] + lv0) * vr0
                    + tanhf(acc[mt][1][r] + lv1) * vr1;
            v += __shfl_xor(v, 1, 32);
            v += __shfl_xor(v, 2, 32);
            v += __shfl_xor(v, 4, 32);
            v += __shfl_xor(v, 8, 32);
            if ((lane & 15) == 0) atomicAdd(&scores[rowbase + r], v);
        }
    }
}

// ---- fused per-batch kernel -------------------------------------------
__global__ void __launch_bounds__(256) fused_kernel(
    const float*    __restrict__ am,    // [B,S,H] fp32
    const uint16_t* __restrict__ Ur,    // [H,H] bf16
    const float*    __restrict__ Lws,   // [B,H] fp32 (precomputed l)
    const float*    __restrict__ Vre,   // [H]
    const float*    __restrict__ Wre,   // [2,H]
    float*          __restrict__ out)   // [B,2]
{
    extern __shared__ char smem[];
    uint16_t* Alds   = (uint16_t*)smem;
    float*    scores = (float*)(smem + A_BYTES);   // 256 floats
    float*    red    = scores + 256;               // 32 floats

    const int tid  = threadIdx.x;
    const int lane = tid & 31;
    const int w    = tid >> 5;
    const int b    = blockIdx.x;

    // Phase 0: stage all_memory[b] -> bf16 LDS slab (208 rows, rows 200..207 zero)
    const float4* src = (const float4*)(am + (size_t)b * Ssz * Hsz);
    for (int i = tid; i < (Ssz * Hsz / 4); i += 256) {
        float4 v = src[i];
        int row  = i >> 7;            // 128 float4 per row
        int colq = (i & 127) << 2;    // half index
        uint32_t* dst = (uint32_t*)(Alds + row * ASTR + colq);
        dst[0] = pack2(v.x, v.y);
        dst[1] = pack2(v.z, v.w);
    }
    {
        uint32_t* Az = (uint32_t*)(Alds + 200 * ASTR);
        for (int i = tid; i < (8 * ASTR / 2); i += 256) Az[i] = 0u;
    }
    scores[tid] = 0.f;
    __syncthreads();

    // Phase 1: WMMA GEMM (13 M-tiles x 32 N-tiles), wave job = 4Mx2N register block.
    // Jobs: mg 0..2 have 4 m-tiles (hot, fully unrolled), mg 3 has 1 (ragged tail).
    const float* lrow = Lws + (size_t)b * Hsz;
    for (int j = w; j < 64; j += 8) {
        int mg = j >> 4;
        int ng = j & 15;
        if (mg < 3) gemm_job<4>(Alds, Ur, lrow, Vre, scores, mg, ng, lane);
        else        gemm_job<1>(Alds, Ur, lrow, Vre, scores, mg, ng, lane);
    }
    __syncthreads();

    // Phase 2: softmax over s<200
    float sv = (tid < Ssz) ? scores[tid] : -3.0e38f;
    float m = sv;
#pragma unroll
    for (int o = 16; o >= 1; o >>= 1) m = fmaxf(m, __shfl_xor(m, o, 32));
    if (lane == 0) red[w] = m;
    __syncthreads();
    if (tid == 0) {
        float mm = red[0];
        for (int i = 1; i < 8; ++i) mm = fmaxf(mm, red[i]);
        red[16] = mm;
    }
    __syncthreads();
    float maxv = red[16];
    float e = (tid < Ssz) ? __expf(sv - maxv) : 0.f;
    float s = e;
#pragma unroll
    for (int o = 16; o >= 1; o >>= 1) s += __shfl_xor(s, o, 32);
    if (lane == 0) red[8 + w] = s;
    __syncthreads();
    if (tid == 0) {
        float ss = 0.f;
        for (int i = 0; i < 8; ++i) ss += red[8 + i];
        red[17] = 1.f / ss;
    }
    __syncthreads();
    float inv = red[17];
    scores[tid] = e * inv;            // attn weights (0 for padded rows)
    __syncthreads();

    // Phase 3: ctx = sum_s attn[s]*all_memory[b,s,:]  (thread owns cols 2t,2t+1)
    float c0 = 0.f, c1 = 0.f;
    int col2 = tid << 1;
    for (int si = 0; si < Ssz; ++si) {
        float wgt = scores[si];
        uint32_t pk = *(const uint32_t*)(Alds + si * ASTR + col2);
        c0 = fmaf(wgt, bf2f(pk & 0xffffu), c0);
        c1 = fmaf(wgt, bf2f(pk >> 16),     c1);
    }
    // logits + 2-way softmax
    float g0 = c0 * Wre[col2]       + c1 * Wre[col2 + 1];
    float g1 = c0 * Wre[Hsz + col2] + c1 * Wre[Hsz + col2 + 1];
#pragma unroll
    for (int o = 16; o >= 1; o >>= 1) { g0 += __shfl_xor(g0, o, 32); g1 += __shfl_xor(g1, o, 32); }
    __syncthreads();
    if (lane == 0) { red[w] = g0; red[8 + w] = g1; }
    __syncthreads();
    if (tid == 0) {
        float t0 = 0.f, t1 = 0.f;
        for (int i = 0; i < 8; ++i) { t0 += red[i]; t1 += red[8 + i]; }
        float mx = fmaxf(t0, t1);
        float e0 = __expf(t0 - mx), e1 = __expf(t1 - mx);
        float is = 1.f / (e0 + e1);
        out[2 * b]     = e0 * is;
        out[2 * b + 1] = e1 * is;
    }
}

// ---- host launch -------------------------------------------------------
extern "C" void kernel_launch(void* const* d_in, const int* in_sizes, int n_in,
                              void* d_out, int out_size, void* d_ws, size_t ws_size,
                              hipStream_t stream) {
    const float* am  = (const float*)d_in[0];   // [B,S,H]
    const float* lm  = (const float*)d_in[1];   // [B,H]
    const float* Urw = (const float*)d_in[2];   // [H,H]
    const float* Wrw = (const float*)d_in[3];   // [H,H]
    const float* Vre = (const float*)d_in[4];   // [1,H]
    const float* Wre = (const float*)d_in[5];   // [2,H]
    float* out = (float*)d_out;

    char* ws = (char*)d_ws;
    uint16_t* Ur16 = (uint16_t*)ws;                                   // 512KB
    uint16_t* Wr16 = (uint16_t*)(ws + 512 * 512 * 2);                 // 512KB
    uint16_t* LM16 = (uint16_t*)(ws + 512 * 512 * 4);                 // 2MB
    float*    Lws  = (float*)(ws + 512 * 512 * 4 + 2048 * 512 * 2);   // 4MB

    convert_bf16<<<512,  256, 0, stream>>>(Urw, Ur16, 512 * 512);
    convert_bf16<<<512,  256, 0, stream>>>(Wrw, Wr16, 512 * 512);
    convert_bf16<<<2048, 256, 0, stream>>>(lm,  LM16, 2048 * 512);
    lgemm_kernel<<<1024, 32, 0, stream>>>(LM16, Wr16, Lws);

    hipFuncSetAttribute((const void*)fused_kernel,
                        hipFuncAttributeMaxDynamicSharedMemorySize, SMEM_BYTES);
    fused_kernel<<<Bsz, 256, SMEM_BYTES, stream>>>(am, Ur16, Lws, Vre, Wre, out);
}
